// CausalSelfAttention_88897233092658
// MI455X (gfx1250) — matrix operand order repaired and checked by reading, not compiled, not run on hardware
//
#include <hip/hip_runtime.h>
#include <hip/hip_bf16.h>
#include <math.h>

// ---------------------------------------------------------------------------
// CDNA5 (gfx1250) causal self-attention:
//   prep (f32->bf16 convert + weight transpose)
//   -> QKV GEMM (WMMA bf16, async global->LDS staging)
//   -> flash attention (WMMA bf16)
//   -> out GEMM (WMMA bf16, async staging)
// ---------------------------------------------------------------------------

typedef __attribute__((ext_vector_type(16))) __bf16 v16bf;
typedef __attribute__((ext_vector_type(8)))  __bf16 v8bf;
typedef __attribute__((ext_vector_type(8)))  float  v8f;

#define BDIM   4
#define HDIM   16
#define TDIM   2048
#define HS     64
#define CDIM   1024

__device__ __forceinline__ v8f wmma_bf16(v16bf a, v16bf b, v8f c) {
    return __builtin_amdgcn_wmma_f32_16x16x32_bf16(
        /*neg_a=*/false, a, /*neg_b=*/false, b,
        /*c_mod=*/(short)0, c, /*reuse_a=*/false, /*reuse_b=*/false);
}

__device__ __forceinline__ v8bf ld8(const __bf16* p) { return *(const v8bf*)p; }

__device__ __forceinline__ v16bf cat16(v8bf a, v8bf b) {
    return __builtin_shufflevector(a, b, 0,1,2,3,4,5,6,7,8,9,10,11,12,13,14,15);
}

// Async global->LDS copy (ASYNCcnt-tracked, 16B per lane).
__device__ __forceinline__ void async_ld_b128(unsigned lds_off, const void* gp) {
    asm volatile("global_load_async_to_lds_b128 %0, %1, off"
                 :: "v"(lds_off), "v"((unsigned long long)(size_t)gp)
                 : "memory");
}
__device__ __forceinline__ void wait_async0() {
    asm volatile("s_wait_asynccnt 0" ::: "memory");
}

// Drain DS ops so cross-lane LDS store->load within a wave is safe,
// without a workgroup barrier (waves run divergent causal loops).
__device__ __forceinline__ void lds_fence_wave() {
    __builtin_amdgcn_wave_barrier();
    asm volatile("s_wait_dscnt 0" ::: "memory");
    __builtin_amdgcn_wave_barrier();
}

__device__ __forceinline__ float rmax16(float v) {
    #pragma unroll
    for (int off = 8; off; off >>= 1) v = fmaxf(v, __shfl_xor(v, off, 16));
    return v;
}
__device__ __forceinline__ float rsum16(float v) {
    #pragma unroll
    for (int off = 8; off; off >>= 1) v += __shfl_xor(v, off, 16);
    return v;
}

// ---------------------------------------------------------------------------
// Prep: flat f32 -> bf16 convert (x), and tiled transpose+convert (weights).
// ---------------------------------------------------------------------------
__global__ __launch_bounds__(256) void cvt_kernel(
    const float* __restrict__ in, __bf16* __restrict__ outp, int n)
{
    int i = (blockIdx.x * 256 + threadIdx.x) * 8;
    if (i + 8 > n) return;
    float4 a = *(const float4*)(in + i);
    float4 b = *(const float4*)(in + i + 4);
    v8bf o = { (__bf16)a.x, (__bf16)a.y, (__bf16)a.z, (__bf16)a.w,
               (__bf16)b.x, (__bf16)b.y, (__bf16)b.z, (__bf16)b.w };
    *(v8bf*)(outp + i) = o;
}

// out[n][k] = (bf16) in[k][n] ; 64x64 tiles through LDS
__global__ __launch_bounds__(256) void transpose_cvt_kernel(
    const float* __restrict__ in, __bf16* __restrict__ outp, int K, int N)
{
    __shared__ __bf16 t[64 * 80];                 // row stride 160B (16B-aligned)
    const int tid = threadIdx.x;
    const int n0 = blockIdx.x * 64, k0 = blockIdx.y * 64;
    #pragma unroll
    for (int g = tid; g < 1024; g += 256) {
        int kk = g >> 4, nn4 = (g & 15) << 2;
        float4 s4 = *(const float4*)(in + (size_t)(k0 + kk) * N + n0 + nn4);
        t[(nn4 + 0) * 80 + kk] = (__bf16)s4.x;
        t[(nn4 + 1) * 80 + kk] = (__bf16)s4.y;
        t[(nn4 + 2) * 80 + kk] = (__bf16)s4.z;
        t[(nn4 + 3) * 80 + kk] = (__bf16)s4.w;
    }
    __syncthreads();
    #pragma unroll
    for (int g = tid; g < 512; g += 256) {
        int nn = g >> 3, kk8 = (g & 7) << 3;
        *(v8bf*)(outp + (size_t)(n0 + nn) * K + k0 + kk8) = *(const v8bf*)&t[nn * 80 + kk8];
    }
}

// ---------------------------------------------------------------------------
// Tiled GEMM: C[M x N] = A[M x K] * Bt[N x K]^T, all bf16, f32 accumulate.
// Workgroup tile 128(M) x 64(N); 8 waves of 32x32; K-step 32.
// Staging via GLOBAL_LOAD_ASYNC_TO_LDS_B128 (no VGPR round-trip).
// MODE 0: scatter to Q (scaled 1/8), K ([B,H,T,64]) and V^T ([B,H,64,T]) bf16.
// MODE 1: plain f32 store.
// ---------------------------------------------------------------------------
template<int MODE>
__global__ __launch_bounds__(256) void gemm_kernel(
    const __bf16* __restrict__ A, const __bf16* __restrict__ Bt,
    int Kdim, int Ncols,
    __bf16* __restrict__ qbuf, __bf16* __restrict__ kbuf,
    __bf16* __restrict__ vtbuf, float* __restrict__ outp)
{
    constexpr int LDA = 40;   // bf16 elems per LDS row (80B: 16B-aligned frags, bank spread)
    constexpr int LDB = 40;
    __shared__ __bf16 As[128 * LDA];
    __shared__ __bf16 Bs[64 * LDB];

    const int tid  = threadIdx.x;
    const int wid  = tid >> 5;
    const int lane = tid & 31;
    const int lm   = lane & 15;
    const int lh   = lane >> 4;
    const int wm   = wid & 3;    // wave M position (4 x 32 rows)
    const int wn   = wid >> 2;   // wave N position (2 x 32 cols)
    const int m0   = blockIdx.y * 128;
    const int n0   = blockIdx.x * 64;

    const unsigned asb = (unsigned)(size_t)&As[0];
    const unsigned bsb = (unsigned)(size_t)&Bs[0];

    v8f acc[2][2] = {};

    for (int kb = 0; kb < Kdim; kb += 32) {
        // ---- async stage A (128x32) and B (64x32, already [n][k]) into LDS ----
        {
            int s0 = tid;                        // A slot 0
            int r  = s0 >> 2, c8 = (s0 & 3) << 3;
            async_ld_b128(asb + (unsigned)(r * LDA + c8) * 2,
                          A + (size_t)(m0 + r) * Kdim + kb + c8);
            int s1 = tid + 256;                  // A slot 1
            r = s1 >> 2; c8 = (s1 & 3) << 3;
            async_ld_b128(asb + (unsigned)(r * LDA + c8) * 2,
                          A + (size_t)(m0 + r) * Kdim + kb + c8);
            int n  = tid >> 2, ck = (tid & 3) << 3;   // B slot
            async_ld_b128(bsb + (unsigned)(n * LDB + ck) * 2,
                          Bt + (size_t)(n0 + n) * Kdim + kb + ck);
        }
        wait_async0();
        __syncthreads();

        // ---- fragments + WMMA ----
        v16bf afr[2], bfr[2];
        #pragma unroll
        for (int mt = 0; mt < 2; mt++) {
            const __bf16* p = &As[(wm * 32 + mt * 16 + lm) * LDA + 8 * lh];
            afr[mt] = cat16(ld8(p), ld8(p + 16));
        }
        #pragma unroll
        for (int nt = 0; nt < 2; nt++) {
            const __bf16* q = &Bs[(wn * 32 + nt * 16 + lm) * LDB + 8 * lh];
            bfr[nt] = cat16(ld8(q), ld8(q + 16));
        }
        #pragma unroll
        for (int mt = 0; mt < 2; mt++)
            #pragma unroll
            for (int nt = 0; nt < 2; nt++)
                acc[mt][nt] = wmma_bf16(afr[mt], bfr[nt], acc[mt][nt]);
        __syncthreads();
    }

    // ---- epilogue ----
    if (MODE == 0) {
        const int which = n0 >> 10;            // 0=Q 1=K 2=V (64-wide tiles never straddle)
        const int head  = (n0 & 1023) >> 6;
        #pragma unroll
        for (int mt = 0; mt < 2; mt++) {
            #pragma unroll
            for (int nt = 0; nt < 2; nt++) {
                #pragma unroll
                for (int v = 0; v < 8; v++) {
                    int r = m0 + wm * 32 + mt * 16 + v + 8 * lh;
                    int b = r >> 11, t = r & 2047;
                    int d = wn * 32 + nt * 16 + lm;
                    float val = acc[mt][nt][v];
                    if (which == 0)
                        qbuf[(((size_t)b * HDIM + head) * TDIM + t) * HS + d] =
                            (__bf16)(val * 0.125f);     // 1/sqrt(64) folded into Q
                    else if (which == 1)
                        kbuf[(((size_t)b * HDIM + head) * TDIM + t) * HS + d] = (__bf16)val;
                    else
                        vtbuf[(((size_t)b * HDIM + head) * HS + d) * TDIM + t] = (__bf16)val;
                }
            }
        }
    } else {
        #pragma unroll
        for (int mt = 0; mt < 2; mt++)
            #pragma unroll
            for (int nt = 0; nt < 2; nt++)
                #pragma unroll
                for (int v = 0; v < 8; v++) {
                    int r = m0 + wm * 32 + mt * 16 + v + 8 * lh;
                    outp[(size_t)r * Ncols + n0 + wn * 32 + nt * 16 + lm] = acc[mt][nt][v];
                }
    }
}

// ---------------------------------------------------------------------------
// Flash attention: one wave = 16 query rows; 32-key blocks; online softmax.
// Q,K: [B*H, T, 64] bf16 (Q pre-scaled). Vt: [B*H, 64, T] bf16. Y: [B,T,C] bf16.
// ---------------------------------------------------------------------------
__global__ __launch_bounds__(256) void attn_kernel(
    const __bf16* __restrict__ Q, const __bf16* __restrict__ Kb,
    const __bf16* __restrict__ Vt, __bf16* __restrict__ Y)
{
    constexpr int LDP = 40;
    __shared__ __bf16 Ps[8 * 16 * LDP];   // per-wave P relayout scratch

    const int tid  = threadIdx.x;
    const int wid  = tid >> 5;
    const int lane = tid & 31;
    const int lm   = lane & 15;
    const int lh   = lane >> 4;
    const int bh   = blockIdx.x >> 4;                         // 0..63
    const int q0   = ((blockIdx.x & 15) * 8 + wid) * 16;      // query tile base
    __bf16* myP = &Ps[wid * 16 * LDP];

    // Q A-fragments for d-chunks [0,32) and [32,64), kept in registers.
    const __bf16* qp = Q + ((size_t)bh * TDIM + q0 + lm) * HS;
    v16bf qa0 = cat16(ld8(qp +      8 * lh), ld8(qp + 16 + 8 * lh));
    v16bf qa1 = cat16(ld8(qp + 32 + 8 * lh), ld8(qp + 48 + 8 * lh));

    v8f o[4] = {};
    float mrow[8], lrow[8];
    #pragma unroll
    for (int v = 0; v < 8; v++) { mrow[v] = -INFINITY; lrow[v] = 0.f; }

    const v8f zero = {};
    const int kend = q0 + 16;                 // need keys <= q0+15 (causal)

    for (int kb = 0; kb < kend; kb += 32) {
        // ---- S = Q * K^T for 32 keys (2 n-tiles x 2 d-chunks) ----
        v8f s[2];
        #pragma unroll
        for (int j = 0; j < 2; j++) {
            const __bf16* kpj = Kb + ((size_t)bh * TDIM + kb + j * 16 + lm) * HS;
            v16bf b0 = cat16(ld8(kpj +      8 * lh), ld8(kpj + 16 + 8 * lh));
            v16bf b1 = cat16(ld8(kpj + 32 + 8 * lh), ld8(kpj + 48 + 8 * lh));
            v8f sj = wmma_bf16(qa0, b0, zero);
            s[j]   = wmma_bf16(qa1, b1, sj);
        }

        // ---- causal mask + online softmax (row = v + 8*lh, col across lanes) ----
        float alpha[8];
        #pragma unroll
        for (int v = 0; v < 8; v++) {
            int qg  = q0 + v + 8 * lh;
            float s0v = (kb + lm      <= qg) ? s[0][v] : -INFINITY;
            float s1v = (kb + 16 + lm <= qg) ? s[1][v] : -INFINITY;
            float mn  = fmaxf(mrow[v], rmax16(fmaxf(s0v, s1v)));
            alpha[v]  = __expf(mrow[v] - mn);
            mrow[v]   = mn;
            float p0 = __expf(s0v - mn);
            float p1 = __expf(s1v - mn);
            lrow[v]  = lrow[v] * alpha[v] + rsum16(p0 + p1);
            myP[(v + 8 * lh) * LDP + lm]      = (__bf16)p0;
            myP[(v + 8 * lh) * LDP + 16 + lm] = (__bf16)p1;
            #pragma unroll
            for (int nt = 0; nt < 4; nt++) o[nt][v] *= alpha[v];
        }

        lds_fence_wave();

        // ---- re-read P in A-fragment layout ----
        const __bf16* pp = &myP[lm * LDP];
        v16bf pa = cat16(ld8(pp + 8 * lh), ld8(pp + 16 + 8 * lh));

        lds_fence_wave();   // reads done before next iteration's stores

        // ---- O += P * V  (V^T rows are contiguous in keys) ----
        const __bf16* vp = Vt + (size_t)bh * HS * TDIM;
        #pragma unroll
        for (int nt = 0; nt < 4; nt++) {
            const __bf16* vq = vp + (size_t)(nt * 16 + lm) * TDIM + kb + 8 * lh;
            v16bf vb = cat16(ld8(vq), ld8(vq + 16));
            o[nt] = wmma_bf16(pa, vb, o[nt]);
        }
    }

    // ---- normalize + store y as bf16 [B,T,C] ----
    const int b = bh >> 4, head = bh & 15;
    #pragma unroll
    for (int v = 0; v < 8; v++) {
        float inv = 1.0f / lrow[v];
        int t = q0 + v + 8 * lh;
        #pragma unroll
        for (int nt = 0; nt < 4; nt++) {
            int c = head * HS + nt * 16 + lm;
            Y[((size_t)b * TDIM + t) * CDIM + c] = (__bf16)(o[nt][v] * inv);
        }
    }
}

// ---------------------------------------------------------------------------
extern "C" void kernel_launch(void* const* d_in, const int* in_sizes, int n_in,
                              void* d_out, int out_size, void* d_ws, size_t ws_size,
                              hipStream_t stream) {
    (void)in_sizes; (void)n_in; (void)out_size; (void)ws_size;
    const float* x      = (const float*)d_in[0];   // [4,2048,1024]
    const float* w_attn = (const float*)d_in[1];   // [1024,3072]
    const float* w_o    = (const float*)d_in[2];   // [1024,1024]
    float* out = (float*)d_out;                    // [4,2048,1024]

    const size_t NX  = (size_t)BDIM * TDIM * CDIM;          // 8,388,608
    const size_t NWA = (size_t)CDIM * 3 * CDIM;             // 3,145,728
    const size_t NWO = (size_t)CDIM * CDIM;                 // 1,048,576
    const size_t NQ  = (size_t)BDIM * HDIM * TDIM * HS;     // 8,388,608

    __bf16* xb    = (__bf16*)d_ws;
    __bf16* watT  = xb + NX;        // [3C, C]   (w_attn^T)
    __bf16* woT   = watT + NWA;     // [C, C]    (w_o^T)
    __bf16* qbuf  = woT + NWO;
    __bf16* kbuf  = qbuf + NQ;
    __bf16* vtbuf = kbuf + NQ;
    __bf16* ybuf  = vtbuf + NQ;     // [B,T,C] bf16

    dim3 blk(256);
    // Prep: converts + weight transposes (one-time, L2-resident afterwards)
    cvt_kernel<<<dim3((unsigned)(NX / 2048)), blk, 0, stream>>>(x, xb, (int)NX);
    transpose_cvt_kernel<<<dim3(3 * CDIM / 64, CDIM / 64), blk, 0, stream>>>(
        w_attn, watT, CDIM, 3 * CDIM);
    transpose_cvt_kernel<<<dim3(CDIM / 64, CDIM / 64), blk, 0, stream>>>(
        w_o, woT, CDIM, CDIM);

    // Stage 1: QKV projection  M=8192 N=3072 K=1024
    gemm_kernel<0><<<dim3(3 * CDIM / 64, 8192 / 128), blk, 0, stream>>>(
        xb, watT, CDIM, 3 * CDIM, qbuf, kbuf, vtbuf, nullptr);
    // Stage 2: causal flash attention (64 heads x 16 q-blocks)
    attn_kernel<<<dim3(64 * 16), blk, 0, stream>>>(qbuf, kbuf, vtbuf, ybuf);
    // Stage 3: output projection  M=8192 N=1024 K=1024
    gemm_kernel<1><<<dim3(CDIM / 64, 8192 / 128), blk, 0, stream>>>(
        ybuf, woT, CDIM, CDIM, nullptr, nullptr, nullptr, out);
}
